// GenClassifier_72782515798535
// MI455X (gfx1250) — matrix operand order repaired and checked by reading, not compiled, not run on hardware
//
#include <hip/hip_runtime.h>

typedef __attribute__((ext_vector_type(16))) _Float16 v16h;
typedef __attribute__((ext_vector_type(8)))  float    v8f;
typedef __attribute__((ext_vector_type(4)))  float    v4f;

#define B_     256
#define C_     3
#define HW_    16384
#define CHW_   49152
#define NC_    10
#define NP_    16
#define KCH_   4          // MH chain steps
#define NS_    5          // states: s0 + 4 candidates
#define PSPLIT 32
#define CHUNK  (HW_ / PSPLIT)   // 512 pixels per workgroup
#define TILES  (CHUNK / 32)     // 16 K-tiles of 32
#define NWAVE  8

// ---------------------------------------------------------------------------
// Zero the logits accumulator in workspace (d_ws is poisoned by the harness).
// ---------------------------------------------------------------------------
__global__ void zero_kernel(float* __restrict__ p, int n) {
    int i = blockIdx.x * 256 + threadIdx.x;
    if (i < n) p[i] = 0.0f;
}

// ---------------------------------------------------------------------------
// logits[s][b][n] = sum_{c,p} x[b,c,p] * z_s[b,p] * W[c*HW+p, n]   (no bias)
// WMMA f32<-f16: M=16 b-rows, N=16 (10 real cols), K accumulated 32 at a time.
// ---------------------------------------------------------------------------
__global__ __launch_bounds__(256) void logits_kernel(
    const float* __restrict__ zl,    // [B,1,H,W]
    const float* __restrict__ x,     // [B,C,H,W]
    const float* __restrict__ bern,  // [K,B,1,H,W]
    const float* __restrict__ W,     // [CHW, NC]
    float* __restrict__ glog)        // [NS, B, NP] accumulator (pre-zeroed)
{
    // LDS: phase 1 = f16 W tiles (blocked [bi][n][k], 48KB); phase 2 = f32 reduce
    __shared__ __align__(64) unsigned char smem[C_ * CHUNK * NP_ * 2]; // 49152B
    _Float16* wlds = (_Float16*)smem;
    float*    red  = (float*)smem;

    const int tid  = threadIdx.x;
    const int b0   = blockIdx.x * 16;       // 16 batch rows per workgroup
    const int p0c  = blockIdx.y * CHUNK;    // pixel chunk base

    // ---- stage W (f32 -> f16, zero-pad cols 10..15) into blocked LDS layout
    for (int i = tid; i < C_ * CHUNK * NP_; i += 256) {
        int n = i & 15;
        int p = (i >> 4) & (CHUNK - 1);
        int c = i >> (4 + 9);                         // log2(CHUNK)==9
        float v = (n < NC_) ? W[(size_t)(c * HW_ + p0c + p) * NC_ + n] : 0.0f;
        int bi = c * TILES + (p >> 5);
        int k  = ((p & 7) | (((p >> 4) & 1) << 3) | (((p >> 3) & 1) << 4));
        wlds[bi * 512 + n * 32 + k] = (_Float16)v;    // [bi][n(16)][k(32), k bits 3,4 swapped]
    }
    __syncthreads();

    const int lane = tid & 31;
    const int wv   = tid >> 5;
    const int col  = lane & 15;     // A-matrix row index AND C/D column index
    const int half = lane >> 4;
    const int b    = b0 + col;      // this lane's batch row for A loads
    const int base = half * 8;      // K-offset base per 16-bit A layout

    v8f acc[NS_] = {};              // 5 state accumulators, f32 16x16 tiles

    #pragma unroll
    for (int t = wv; t < TILES; t += NWAVE) {
        const int pa = p0c + t * 32 + base; // run1: pa..pa+7, run2: pa+16..pa+23

        // ---- z masks for 5 states at this lane's 16 K positions.
        // bern < sigmoid(zl)  <=>  bern*(1+exp(-zl)) < 1   (division-free)
        const float* zlp = zl + (size_t)b * HW_ + pa;
        __builtin_prefetch(zlp + NWAVE * 32, 0, 1);   // global_prefetch_b8
        float zv[16] __attribute__((aligned(32)));
        *(v4f*)&zv[0]  = *(const v4f*)(zlp + 0);
        *(v4f*)&zv[4]  = *(const v4f*)(zlp + 4);
        *(v4f*)&zv[8]  = *(const v4f*)(zlp + 16);
        *(v4f*)&zv[12] = *(const v4f*)(zlp + 20);

        float tinv[16];                                // 1 + exp(-zl)
        #pragma unroll
        for (int i = 0; i < 16; ++i) tinv[i] = 1.0f + __expf(-zv[i]);

        v16h msk[NS_];
        #pragma unroll
        for (int i = 0; i < 16; ++i)   // greedy state: sigmoid(zl)>=0.5 <=> zl>=0
            msk[0][i] = (zv[i] >= 0.0f) ? (_Float16)1.0f : (_Float16)0.0f;

        #pragma unroll
        for (int s = 1; s < NS_; ++s) {
            const float* bp = bern + ((size_t)(s - 1) * B_ + b) * HW_ + pa;
            __builtin_prefetch(bp + NWAVE * 32, 0, 1);
            float bv[16] __attribute__((aligned(32)));
            *(v4f*)&bv[0]  = *(const v4f*)(bp + 0);
            *(v4f*)&bv[4]  = *(const v4f*)(bp + 4);
            *(v4f*)&bv[8]  = *(const v4f*)(bp + 16);
            *(v4f*)&bv[12] = *(const v4f*)(bp + 20);
            #pragma unroll
            for (int i = 0; i < 16; ++i)
                msk[s][i] = (bv[i] * tinv[i] < 1.0f) ? (_Float16)1.0f
                                                     : (_Float16)0.0f;
        }

        // ---- K accumulation over channels: 5 WMMAs per channel
        #pragma unroll
        for (int c = 0; c < C_; ++c) {
            const float* xp = x + (size_t)b * CHW_ + (size_t)c * HW_ + pa;
            __builtin_prefetch(xp + NWAVE * 32, 0, 1);
            float xv[16] __attribute__((aligned(32)));
            *(v4f*)&xv[0]  = *(const v4f*)(xp + 0);
            *(v4f*)&xv[4]  = *(const v4f*)(xp + 4);
            *(v4f*)&xv[8]  = *(const v4f*)(xp + 16);
            *(v4f*)&xv[12] = *(const v4f*)(xp + 20);
            v16h xh;
            #pragma unroll
            for (int i = 0; i < 16; ++i) xh[i] = (_Float16)xv[i];

            // B operand: lane (n=col, K half) -> 16 contiguous f16 in LDS
            const v16h bw = *(const v16h*)&wlds[(c * TILES + t) * 512 +
                                                col * 32 + half * 16];
            #pragma unroll
            for (int s = 0; s < NS_; ++s) {
                v16h a = xh * msk[s];  // masked A tile (v_pk_mul_f16)
                acc[s] = __builtin_amdgcn_wmma_f32_16x16x32_f16(
                    false, a, false, bw, (short)0, acc[s], false, false);
            }
        }
    }

    // ---- reduce 8 waves through LDS, then one atomic pass to global
    __syncthreads();                              // done with wlds
    for (int i = tid; i < NS_ * 16 * NP_; i += 256) red[i] = 0.0f;
    __syncthreads();
    #pragma unroll
    for (int s = 0; s < NS_; ++s)
        #pragma unroll
        for (int i = 0; i < 8; ++i)               // C/D: row = i + 8*half, col = lane&15
            atomicAdd(&red[(s * 16 + (i + half * 8)) * NP_ + col], acc[s][i]);
    __syncthreads();
    for (int i = tid; i < NS_ * 16 * NP_; i += 256) {
        int n  = i & 15;
        int rm = (i >> 4) & 15;
        int s  = i >> 8;
        atomicAdd(&glog[((size_t)s * B_ + b0 + rm) * NP_ + n], red[i]);
    }
}

// ---------------------------------------------------------------------------
// Per batch row: log-softmax -> pz[s], 4-step MH chain, emit logits[idx]+bias.
// ---------------------------------------------------------------------------
__global__ void finalize_kernel(const float* __restrict__ glog,
                                const int*   __restrict__ y,
                                const float* __restrict__ u,     // [K,B]
                                const float* __restrict__ bias,  // [NC]
                                float* __restrict__ out)         // [B,NC]
{
    int b = blockIdx.x * blockDim.x + threadIdx.x;
    if (b >= B_) return;

    float pz[NS_];
    #pragma unroll
    for (int s = 0; s < NS_; ++s) {
        float l[NC_], mx = -1e30f;
        #pragma unroll
        for (int n = 0; n < NC_; ++n) {
            l[n] = glog[((size_t)s * B_ + b) * NP_ + n] + bias[n];
            mx = fmaxf(mx, l[n]);
        }
        float sum = 0.0f;
        #pragma unroll
        for (int n = 0; n < NC_; ++n) sum += __expf(l[n] - mx);
        float lse = mx + __logf(sum);
        pz[s] = l[y[b]] - lse;
    }

    int idx = 0;
    float lp = pz[0];
    #pragma unroll
    for (int k = 1; k <= KCH_; ++k) {
        float lu = __logf(u[(k - 1) * B_ + b]);
        if (lu <= pz[k] - lp) { idx = k; lp = pz[k]; }
    }

    #pragma unroll
    for (int n = 0; n < NC_; ++n)
        out[b * NC_ + n] = glog[((size_t)idx * B_ + b) * NP_ + n] + bias[n];
}

// ---------------------------------------------------------------------------
extern "C" void kernel_launch(void* const* d_in, const int* in_sizes, int n_in,
                              void* d_out, int out_size, void* d_ws, size_t ws_size,
                              hipStream_t stream) {
    const float* zl   = (const float*)d_in[0];
    const float* x    = (const float*)d_in[1];
    const int*   y    = (const int*)  d_in[2];
    const float* bern = (const float*)d_in[3];
    const float* u    = (const float*)d_in[4];
    const float* W    = (const float*)d_in[5];
    const float* bias = (const float*)d_in[6];
    float* out  = (float*)d_out;
    float* glog = (float*)d_ws;                   // NS*B*NP f32 = 80KB

    const int nlog = NS_ * B_ * NP_;
    zero_kernel<<<(nlog + 255) / 256, 256, 0, stream>>>(glog, nlog);
    logits_kernel<<<dim3(B_ / 16, PSPLIT), 256, 0, stream>>>(zl, x, bern, W, glog);
    finalize_kernel<<<1, 256, 0, stream>>>(glog, y, u, bias, out);
}
